// encoder_34497177322219
// MI455X (gfx1250) — matrix operand order repaired and checked by reading, not compiled, not run on hardware
//
#include <hip/hip_runtime.h>

// ---------------- problem constants ----------------
#define NN   20000      // nodes
#define NE   160000     // edges
#define CIN  2000
#define K1P  2048       // CIN padded to mult of 32 (WMMA K)
#define N1   500
#define N1P  512        // padded N1 (GEMM1 out cols / GEMM2 K)
#define N2   250
#define N2P  256        // padded N2

typedef __attribute__((ext_vector_type(16))) __bf16 v16bf;
typedef __attribute__((ext_vector_type(8)))  float  v8f;

union Frag { v16bf v; uint4 q[2]; };

__device__ __forceinline__ unsigned short f32_to_bf16(float f) {
  unsigned int u = __float_as_uint(f);
  u += 0x7fffu + ((u >> 16) & 1u);   // round-to-nearest-even
  return (unsigned short)(u >> 16);
}

__device__ __forceinline__ float softplus_f(float x) {
  return fmaxf(x, 0.f) + log1pf(expf(-fabsf(x)));
}

// ---------------- small prep kernels ----------------
__global__ void k_fill_deg(float* deg) {               // self-loop weight 1
  int i = blockIdx.x * blockDim.x + threadIdx.x;
  if (i < NN) deg[i] = 1.0f;
}
__global__ void k_deg_edges(const int* __restrict__ dst, const float* __restrict__ ew,
                            float* __restrict__ deg) {
  int e = blockIdx.x * blockDim.x + threadIdx.x;
  if (e < NE) atomicAdd(&deg[dst[e]], ew[e]);
}
__global__ void k_dinv(const float* __restrict__ deg, float* __restrict__ dinv) {
  int i = blockIdx.x * blockDim.x + threadIdx.x;
  if (i < NN) { float d = deg[i]; dinv[i] = d > 0.f ? rsqrtf(d) : 0.f; }
}
__global__ void k_coef(const int* __restrict__ src, const int* __restrict__ dst,
                       const float* __restrict__ ew, const float* __restrict__ dinv,
                       float* __restrict__ coef) {
  int e = blockIdx.x * blockDim.x + threadIdx.x;
  if (e < NE) coef[e] = dinv[src[e]] * ew[e] * dinv[dst[e]];
}

// X [NN,CIN] fp32 -> [NN,K1P] bf16 (zero-pad K)
__global__ void k_cvt_x(const float* __restrict__ x, unsigned short* __restrict__ xb) {
  int t = blockIdx.x * blockDim.x + threadIdx.x;
  if (t >= NN * K1P) return;
  int n = t >> 11, k = t & 2047;
  float v = (k < CIN) ? x[(size_t)n * CIN + k] : 0.f;
  xb[t] = f32_to_bf16(v);
}
// generic weight pad+convert: [R,C] fp32 -> [RP,CP] bf16
__global__ void k_cvt_w(const float* __restrict__ w, unsigned short* __restrict__ wb,
                        int R, int C, int CP, int total) {
  int t = blockIdx.x * blockDim.x + threadIdx.x;
  if (t >= total) return;
  int r = t / CP, c = t % CP;
  float v = (r < R && c < C) ? w[(size_t)r * C + c] : 0.f;
  wb[t] = f32_to_bf16(v);
}
__global__ void k_cvt_f32bf16(const float* __restrict__ in, unsigned short* __restrict__ out, int n) {
  int t = blockIdx.x * blockDim.x + threadIdx.x;
  if (t < n) out[t] = f32_to_bf16(in[t]);
}

// ---------------- WMMA bf16 GEMM:  C[M,NG*128] = A[M,KPAD] * B[NG*128,KPAD]^T ----------------
// A,B are K-major bf16 (row = contiguous K).  Wave tile: 32(M) x 128(N).
template<int KPAD, int NG>
__global__ __launch_bounds__(256) void gemm_bf16_wmma(const unsigned short* __restrict__ A,
                                                      const unsigned short* __restrict__ B,
                                                      float* __restrict__ C, int Mtiles) {
  const int lane = threadIdx.x & 31;
  const int wave = blockIdx.x * (blockDim.x >> 5) + (threadIdx.x >> 5);
  if (wave >= Mtiles * NG) return;             // wave-uniform: EXEC all-1s inside
  const int mt = wave / NG, ng = wave % NG;
  const int row0 = mt * 32, col0 = ng * 128;
  const int lr = lane & 15;                    // row/col within tile
  const int hi = lane >> 4;                    // K-half selector

  // A frag (16x32 bf16): lanes0-15 K chunks [0,8)+[16,24); lanes16-31 [8,16)+[24,32)
  const unsigned short* aptr0 = A + (size_t)(row0 + lr) * KPAD + hi * 8;
  const unsigned short* aptr1 = aptr0 + (size_t)16 * KPAD;
  // B frag (32x16 bf16): lanes0-15 K chunks [0,8)+[16,24); lanes16-31 [8,16)+[24,32); col = lr (rows of W)
  const unsigned short* bptr  = B + (size_t)(col0 + lr) * KPAD + hi * 8;

  v8f zero; 
  #pragma unroll
  for (int r = 0; r < 8; ++r) zero[r] = 0.f;
  v8f acc[2][8];
  #pragma unroll
  for (int i = 0; i < 2; ++i)
    #pragma unroll
    for (int j = 0; j < 8; ++j) acc[i][j] = zero;

  #pragma unroll 1
  for (int kb = 0; kb < KPAD; kb += 32) {
    __builtin_prefetch(aptr0 + 32, 0, 0);      // global_prefetch_b8: next K-block of A
    __builtin_prefetch(aptr1 + 32, 0, 0);
    Frag a0, a1;
    a0.q[0] = *(const uint4*)(aptr0);
    a0.q[1] = *(const uint4*)(aptr0 + 16);
    a1.q[0] = *(const uint4*)(aptr1);
    a1.q[1] = *(const uint4*)(aptr1 + 16);
    #pragma unroll
    for (int nt = 0; nt < 8; ++nt) {
      Frag bf;
      const unsigned short* bp = bptr + (size_t)nt * 16 * KPAD;
      bf.q[0] = *(const uint4*)(bp);
      bf.q[1] = *(const uint4*)(bp + 16);
      acc[0][nt] = __builtin_amdgcn_wmma_f32_16x16x32_bf16(false, a0.v, false, bf.v,
                                                           (short)0, acc[0][nt], false, false);
      acc[1][nt] = __builtin_amdgcn_wmma_f32_16x16x32_bf16(false, a1.v, false, bf.v,
                                                           (short)0, acc[1][nt], false, false);
    }
    aptr0 += 32; aptr1 += 32; bptr += 32;
  }

  const int NPAD = NG * 128;
  #pragma unroll
  for (int i = 0; i < 2; ++i) {
    const int rbase = row0 + i * 16 + hi * 8;  // C layout: VGPR r -> row rbase+r, col = lr
    #pragma unroll
    for (int nt = 0; nt < 8; ++nt) {
      float* cp = C + (size_t)rbase * NPAD + col0 + nt * 16 + lr;
      #pragma unroll
      for (int r = 0; r < 8; ++r) cp[(size_t)r * NPAD] = acc[i][nt][r];
    }
  }
}

// ---------------- aggregation (GCN scatter) ----------------
// AGG[n,f] = dinv[n]^2 * H[n,f] + b[f]   (self-loop term, pad cols zeroed)
__global__ void k_agg_init(const float* __restrict__ H, const float* __restrict__ dinv,
                           const float* __restrict__ bias, float* __restrict__ AGG,
                           int shift, int F, int total) {
  int t = blockIdx.x * blockDim.x + threadIdx.x;
  if (t >= total) return;
  int n = t >> shift, f = t & ((1 << shift) - 1);
  float v = 0.f;
  if (f < F) { float di = dinv[n]; v = di * di * H[t] + bias[f]; }
  AGG[t] = v;
}
// edge messages: AGG[dst] += coef[e] * H[src]   (fp32 global atomics)
__global__ void k_agg_edges(const float* __restrict__ H, const int* __restrict__ src,
                            const int* __restrict__ dst, const float* __restrict__ coef,
                            float* __restrict__ AGG, int stride, int F) {
  int e = blockIdx.x;
  int s = src[e], d = dst[e];
  float c = coef[e];
  const float* hs = H + (size_t)s * stride;
  float* ad = AGG + (size_t)d * stride;
  for (int f = threadIdx.x; f < F; f += blockDim.x) atomicAdd(&ad[f], c * hs[f]);
}

// ---------------- output heads ----------------
__global__ void k_heads(const float* __restrict__ h, int stride,
                        const float* Wmt, const float* bmt, const float* Wst, const float* bst,
                        const float* Wmz, const float* bmz, const float* Wsz, const float* bsz,
                        float* __restrict__ out) {
  int t = blockIdx.x * blockDim.x + threadIdx.x;
  if (t >= NN * 34) return;
  int node = t / 34, o = t % 34;
  const float* w; float b; bool sp; float* op;
  if (o == 0)      { w = Wmt;                 b = bmt[0];    sp = true;  op = out + node; }
  else if (o == 1) { w = Wst;                 b = bst[0];    sp = true;  op = out + NN + node; }
  else if (o < 18) { int j = o - 2;  w = Wmz + (size_t)j * N2; b = bmz[j]; sp = false;
                     op = out + 2 * NN + (size_t)node * 16 + j; }
  else             { int j = o - 18; w = Wsz + (size_t)j * N2; b = bsz[j]; sp = true;
                     op = out + 2 * NN + 16 * NN + (size_t)node * 16 + j; }
  const float* hr = h + (size_t)node * stride;
  float acc = b;
  #pragma unroll 5
  for (int k = 0; k < N2; ++k) acc = fmaf(hr[k], w[k], acc);
  *op = sp ? softplus_f(acc) : acc;
}

// ---------------- host launch ----------------
static inline int cdiv(int a, int b) { return (a + b - 1) / b; }

extern "C" void kernel_launch(void* const* d_in, const int* in_sizes, int n_in,
                              void* d_out, int out_size, void* d_ws, size_t ws_size,
                              hipStream_t stream) {
  (void)in_sizes; (void)n_in; (void)out_size; (void)ws_size;
  const float* x   = (const float*)d_in[0];
  const int*   ei  = (const int*)  d_in[1];
  const float* ew  = (const float*)d_in[2];
  const float* W1  = (const float*)d_in[3];
  const float* b1  = (const float*)d_in[4];
  const float* W2  = (const float*)d_in[5];
  const float* b2  = (const float*)d_in[6];
  const float* Wmt = (const float*)d_in[7];
  const float* bmt = (const float*)d_in[8];
  const float* Wst = (const float*)d_in[9];
  const float* bst = (const float*)d_in[10];
  const float* Wmz = (const float*)d_in[11];
  const float* bmz = (const float*)d_in[12];
  const float* Wsz = (const float*)d_in[13];
  const float* bsz = (const float*)d_in[14];
  const int* src = ei;
  const int* dst = ei + NE;
  float* out = (float*)d_out;

  // workspace layout (~135 MB peak, lifetime-aliased)
  char* ws = (char*)d_ws;
  float*          deg  = (float*)(ws + 0);                         //  80 KB
  float*          dinv = (float*)(ws + (96 << 10));                //  80 KB
  float*          coef = (float*)(ws + (192 << 10));               // 640 KB
  unsigned short* w1b  = (unsigned short*)(ws + (1 << 20));        // 2.0 MB  [N1P,K1P]
  unsigned short* w2b  = (unsigned short*)(ws + (1 << 20) + 2097152); // 256 KB [N2P,N1P]
  char* regA = ws + (4  << 20);   // 81.92 MB region
  char* regB = ws + (96 << 20);   // 40.96 MB region
  unsigned short* xb   = (unsigned short*)regA;              // [NN,K1P] bf16 (dies after GEMM1)
  float*          H1   = (float*)regB;                       // [NN,N1P] f32  (dies after agg1)
  float*          AGG1 = (float*)regA;                       // [NN,N1P] f32  (aliases dead xb)
  unsigned short* a1b  = (unsigned short*)(regA + 40960000); // [NN,N1P] bf16
  float*          H2   = (float*)(regA + 61440000);          // [NN,N2P] f32
  float*          AGG2 = (float*)regB;                       // [NN,N2P] f32  (aliases dead H1)

  // degree / normalization
  k_fill_deg <<<cdiv(NN, 256), 256, 0, stream>>>(deg);
  k_deg_edges<<<cdiv(NE, 256), 256, 0, stream>>>(dst, ew, deg);
  k_dinv     <<<cdiv(NN, 256), 256, 0, stream>>>(deg, dinv);
  k_coef     <<<cdiv(NE, 256), 256, 0, stream>>>(src, dst, ew, dinv, coef);

  // layer 1: convert -> WMMA GEMM -> aggregate
  k_cvt_x<<<cdiv(NN * K1P, 256), 256, 0, stream>>>(x, xb);
  k_cvt_w<<<cdiv(N1P * K1P, 256), 256, 0, stream>>>(W1, w1b, N1, CIN, K1P, N1P * K1P);
  gemm_bf16_wmma<K1P, 4><<<cdiv((NN / 32) * 4, 8), 256, 0, stream>>>(xb, w1b, H1, NN / 32);
  k_agg_init <<<cdiv(NN * N1P, 256), 256, 0, stream>>>(H1, dinv, b1, AGG1, 9, N1, NN * N1P);
  k_agg_edges<<<NE, 128, 0, stream>>>(H1, src, dst, coef, AGG1, N1P, N1);

  // layer 2
  k_cvt_f32bf16<<<cdiv(NN * N1P, 256), 256, 0, stream>>>(AGG1, a1b, NN * N1P);
  k_cvt_w<<<cdiv(N2P * N1P, 256), 256, 0, stream>>>(W2, w2b, N2, N1, N1P, N2P * N1P);
  gemm_bf16_wmma<N1P, 2><<<cdiv((NN / 32) * 2, 8), 256, 0, stream>>>(a1b, w2b, H2, NN / 32);
  k_agg_init <<<cdiv(NN * N2P, 256), 256, 0, stream>>>(H2, dinv, b2, AGG2, 8, N2, NN * N2P);
  k_agg_edges<<<NE, 128, 0, stream>>>(H2, src, dst, coef, AGG2, N2P, N2);

  // heads -> d_out (mu_t | std_t | mu_z | std_z)
  k_heads<<<cdiv(NN * 34, 256), 256, 0, stream>>>(AGG2, N2P,
                                                  Wmt, bmt, Wst, bst, Wmz, bmz, Wsz, bsz, out);
}
